// HetGNN_13915694039575
// MI455X (gfx1250) — matrix operand order repaired and checked by reading, not compiled, not run on hardware
//
#include <hip/hip_runtime.h>
#include <hip/hip_bf16.h>
#include <math.h>

#define B_ 128
#define A_ 32
#define U_ 64
#define D_ 64

typedef __attribute__((ext_vector_type(16))) _Float16 v16h;
typedef __attribute__((ext_vector_type(8)))  _Float16 v8h;
typedef __attribute__((ext_vector_type(8)))  float    v8f;

union V16H { v16h v; v8h h[2]; };

// ---------------------------------------------------------------------------
// Row-index mapping for on-the-fly concat of broadcast node features.
// edge row e = (b*A + a)*U + u
//   mode 0: direct (row = r)         -- edge tensors, node tensors, sums
//   mode 1: AP broadcast (row = r/U)
//   mode 2: UE broadcast (row = b*U + u)
// ---------------------------------------------------------------------------
__device__ __forceinline__ int maprow(int r, int mode, int A, int U) {
    if (mode == 1) return r / U;
    if (mode == 2) { int b = r / (A * U); int u = r - (r / U) * U; return b * U + u; }
    return r;
}

// A-matrix fragment (f16 16x32): per lane, halfs 0..7 = K kb+hi*8.., halfs 8..15 = K kb+16+hi*8..
__device__ __forceinline__ v16h load_fragA(const _Float16* row, int kb, int hi) {
    V16H u;
    u.h[0] = *(const v8h*)(row + kb + hi * 8);
    u.h[1] = *(const v8h*)(row + kb + 16 + hi * 8);
    return u.v;
}

// B-matrix fragment (f16 32x16): lane n = ncol, halfs 0..7 = K kb+hi*8.., halfs 8..15 = K kb+16+hi*8..
__device__ __forceinline__ v16h load_fragB(const _Float16* W, int ldw, int ncol, int kb, int hi) {
    const _Float16* p = W + (size_t)ncol * ldw + kb + hi * 8;
    V16H u;
    u.h[0] = *(const v8h*)(p);
    u.h[1] = *(const v8h*)(p + 16);
    return u.v;
}

// ---------------------------------------------------------------------------
// Fused 3-layer MLP + per-block BN partial stats, f16 raw output.
// out16 = relu(relu(relu(X W1^T + b1) W2^T + b2) W3^T + b3)   (pre-BN "raw")
// X = cat[src0[map0(r)], src1[map1(r)]]  (each 64 f16 cols, total K=128)
// 256 threads = 8 waves; 128 rows/block, 16 rows/wave.
// Weights staged into LDS once per block; inter-layer activations relayout
// through a single per-wave LDS tile (in-order DS makes this safe);
// 32 v_wmma per wave. Column sum/sumsq partials (f32) written per block.
// ---------------------------------------------------------------------------
__global__ __launch_bounds__(256)
void mlp3_wmma(const _Float16* __restrict__ src0, int mode0,
               const _Float16* __restrict__ src1, int mode1,
               const _Float16* __restrict__ W1, const float* __restrict__ b1,
               const _Float16* __restrict__ W2, const float* __restrict__ b2,
               const _Float16* __restrict__ W3, const float* __restrict__ b3,
               _Float16* __restrict__ out16, float* __restrict__ partial,
               int A, int U)
{
    __shared__ __align__(16) _Float16 sH[8][16 * 72];   // per-wave 16x64 tile, 72-half stride
    __shared__ __align__(16) _Float16 sW1[64 * 128];
    __shared__ __align__(16) _Float16 sW2[64 * 64];
    __shared__ __align__(16) _Float16 sW3[64 * 64];
    __shared__ float sS[8][2][64];
    __shared__ float sQ[8][2][64];

    const int tid  = threadIdx.x;
    const int wave = tid >> 5;
    const int lane = tid & 31;
    const int lrow = lane & 15;
    const int hi   = lane >> 4;

    // ---- stage weights to LDS (32 KB, cooperative b128 copy) ----
    {
        const v8h* g1 = (const v8h*)W1; v8h* l1 = (v8h*)sW1;
        for (int i = tid; i < 1024; i += 256) l1[i] = g1[i];
        const v8h* g2 = (const v8h*)W2; v8h* l2 = (v8h*)sW2;
        for (int i = tid; i < 512; i += 256) l2[i] = g2[i];
        const v8h* g3 = (const v8h*)W3; v8h* l3 = (v8h*)sW3;
        for (int i = tid; i < 512; i += 256) l3[i] = g3[i];
    }
    __syncthreads();

    const int r = blockIdx.x * 128 + wave * 16 + lrow;
    const int row0 = maprow(r, mode0, A, U);
    const int row1 = maprow(r, mode1, A, U);
    const _Float16* a0 = src0 + (size_t)row0 * 64;
    const _Float16* a1 = src1 + (size_t)row1 * 64;

    _Float16* myH = &sH[wave][0];

    // ---- layer 1: K = 128 (two 64-half gathered sources) ----
    v16h Af[4];
    Af[0] = load_fragA(a0, 0, hi);
    Af[1] = load_fragA(a0, 32, hi);
    Af[2] = load_fragA(a1, 0, hi);
    Af[3] = load_fragA(a1, 32, hi);
#pragma unroll
    for (int n = 0; n < 4; ++n) {
        v8f acc = {};
#pragma unroll
        for (int c = 0; c < 4; ++c) {
            v16h Bf = load_fragB(sW1, 128, n * 16 + lrow, c * 32, hi);
            acc = __builtin_amdgcn_wmma_f32_16x16x32_f16(false, Af[c], false, Bf,
                                                         (short)0, acc, false, false);
        }
        float bb = b1[n * 16 + lrow];
#pragma unroll
        for (int i = 0; i < 8; ++i) {
            float v = acc[i] + bb; v = v > 0.f ? v : 0.f;
            myH[(hi * 8 + i) * 72 + n * 16 + lrow] = (_Float16)v;
        }
    }

    // ---- layer 2: K = 64; A-fragments fully loaded before overwrite (in-order DS) ----
    {
        const _Float16* arow = myH + lrow * 72;
        v16h A2[2];
        A2[0] = load_fragA(arow, 0, hi);
        A2[1] = load_fragA(arow, 32, hi);
#pragma unroll
        for (int n = 0; n < 4; ++n) {
            v8f acc = {};
#pragma unroll
            for (int f = 0; f < 2; ++f) {
                v16h Bf = load_fragB(sW2, 64, n * 16 + lrow, f * 32, hi);
                acc = __builtin_amdgcn_wmma_f32_16x16x32_f16(false, A2[f], false, Bf,
                                                             (short)0, acc, false, false);
            }
            float bb = b2[n * 16 + lrow];
#pragma unroll
            for (int i = 0; i < 8; ++i) {
                float v = acc[i] + bb; v = v > 0.f ? v : 0.f;
                myH[(hi * 8 + i) * 72 + n * 16 + lrow] = (_Float16)v;
            }
        }
    }

    // ---- layer 3: K = 64; result -> LDS (f16) + fused BN partial stats ----
    {
        const _Float16* arow = myH + lrow * 72;
        v16h A3[2];
        A3[0] = load_fragA(arow, 0, hi);
        A3[1] = load_fragA(arow, 32, hi);
#pragma unroll
        for (int n = 0; n < 4; ++n) {
            v8f acc = {};
#pragma unroll
            for (int f = 0; f < 2; ++f) {
                v16h Bf = load_fragB(sW3, 64, n * 16 + lrow, f * 32, hi);
                acc = __builtin_amdgcn_wmma_f32_16x16x32_f16(false, A3[f], false, Bf,
                                                             (short)0, acc, false, false);
            }
            float bb = b3[n * 16 + lrow];
            float s = 0.f, q = 0.f;
#pragma unroll
            for (int i = 0; i < 8; ++i) {
                float v = acc[i] + bb; v = v > 0.f ? v : 0.f;
                myH[(hi * 8 + i) * 72 + n * 16 + lrow] = (_Float16)v;
                s += v; q += v * v;
            }
            sS[wave][hi][n * 16 + lrow] = s;
            sQ[wave][hi][n * 16 + lrow] = q;
        }
    }

    // ---- coalesced f16 writeback: 4 x b128 per lane (lane covers half a row) ----
    {
        const _Float16* srow = myH + lrow * 72 + hi * 32;
        _Float16* drow = out16 + ((size_t)blockIdx.x * 128 + (size_t)wave * 16 + lrow) * 64 + hi * 32;
#pragma unroll
        for (int c = 0; c < 4; ++c)
            *(v8h*)(drow + c * 8) = *(const v8h*)(srow + c * 8);
    }

    __syncthreads();
    if (tid < 64) {
        float s = 0.f, q = 0.f;
#pragma unroll
        for (int w = 0; w < 8; ++w) {
            s += sS[w][0][tid] + sS[w][1][tid];
            q += sQ[w][0][tid] + sQ[w][1][tid];
        }
        partial[blockIdx.x * 128 + tid]      = s;
        partial[blockIdx.x * 128 + 64 + tid] = q;
    }
}

// ---------------------------------------------------------------------------
// PreLayer linear+relu (K = 1 or 2), f16 raw out + fused per-block BN partials.
// 128 rows per block, 256 threads.
// ---------------------------------------------------------------------------
__global__ void pre1_k(const float* __restrict__ x, int K,
                       const float* __restrict__ W, const float* __restrict__ b,
                       _Float16* __restrict__ raw, float* __restrict__ partial)
{
    int tid = threadIdx.x;
    int col = tid & 63, rg = tid >> 6;
    size_t base = (size_t)blockIdx.x * 128;
    float wv0 = W[col * K];
    float wv1 = (K > 1) ? W[col * K + 1] : 0.f;
    float bb  = b[col];
    float s = 0.f, s2 = 0.f;
    for (int r = rg; r < 128; r += 4) {
        size_t row = base + r;
        float acc = bb + x[row * K] * wv0;
        if (K > 1) acc += x[row * K + 1] * wv1;
        acc = acc > 0.f ? acc : 0.f;
        raw[row * 64 + col] = (_Float16)acc;
        s += acc; s2 += acc * acc;
    }
    __shared__ float sh[256], sh2[256];
    sh[tid] = s; sh2[tid] = s2;
    __syncthreads();
    if (tid < 64) {
        s  = sh[tid]  + sh[tid + 64]  + sh[tid + 128]  + sh[tid + 192];
        s2 = sh2[tid] + sh2[tid + 64] + sh2[tid + 128] + sh2[tid + 192];
        partial[blockIdx.x * 128 + tid]      = s;
        partial[blockIdx.x * 128 + 64 + tid] = s2;
    }
}

// ---------------------------------------------------------------------------
// Deterministic partial combine, wide version: 1024 threads = 8 chunks x 128
// columns; fixed strided order + fixed 8-way combine keeps bitwise determinism
// while cutting the serial dependence chain from nblocks to nblocks/8 + 8.
// stats[j] = sum, stats[64+j] = sumsq
// ---------------------------------------------------------------------------
__global__ __launch_bounds__(1024)
void stats2_k(const float* __restrict__ partial, int nblocks,
              float* __restrict__ stats)
{
    int tid   = threadIdx.x;      // 0..1023
    int col   = tid & 127;        // 0..127
    int chunk = tid >> 7;         // 0..7
    float s = 0.f;
    for (int b = chunk; b < nblocks; b += 8) s += partial[b * 128 + col];
    __shared__ float sh[1024];
    sh[tid] = s;
    __syncthreads();
    if (tid < 128) {
        float t = 0.f;
#pragma unroll
        for (int c = 0; c < 8; ++c) t += sh[c * 128 + tid];
        stats[tid] = t;
    }
}

// ---------------------------------------------------------------------------
// BN normalize from f16 raw: y = (x - m) * rsqrt(v + eps) * gamma + beta -> f16
// ---------------------------------------------------------------------------
__global__ void norm_k(const _Float16* __restrict__ raw, const float* __restrict__ stats,
                       const float* __restrict__ gamma, const float* __restrict__ beta,
                       float invN, _Float16* __restrict__ o16, int total)
{
    int idx = blockIdx.x * 256 + threadIdx.x;
    if (idx >= total) return;
    int j = idx & 63;
    float m   = stats[j] * invN;
    float var = stats[64 + j] * invN - m * m;
    float sc  = rsqrtf(var + 1e-5f) * gamma[j];
    float y   = ((float)raw[idx] - m) * sc + beta[j];
    o16[idx] = (_Float16)y;
}

// normalize-on-the-fly + sum over U for each (b,a): blocks = B*A, 64 threads
__global__ void normsumU_k(const _Float16* __restrict__ raw, const float* __restrict__ stats,
                           const float* __restrict__ gamma, const float* __restrict__ beta,
                           float invN, float* __restrict__ o32, _Float16* __restrict__ o16,
                           int U)
{
    int row = blockIdx.x, j = threadIdx.x;
    float m   = stats[j] * invN;
    float var = stats[64 + j] * invN - m * m;
    float sc  = rsqrtf(var + 1e-5f) * gamma[j];
    float off = beta[j] - m * sc;
    float s = 0.f;
    size_t base = (size_t)row * U * 64 + j;
    for (int u = 0; u < U; ++u) s += (float)raw[base + (size_t)u * 64] * sc + off;
    if (o32) o32[row * 64 + j] = s;
    if (o16) o16[row * 64 + j] = (_Float16)s;
}

// normalize-on-the-fly + sum over A for each (b,u): blocks = B*U, 64 threads
__global__ void normsumA_k(const _Float16* __restrict__ raw, const float* __restrict__ stats,
                           const float* __restrict__ gamma, const float* __restrict__ beta,
                           float invN, float* __restrict__ o32, _Float16* __restrict__ o16,
                           int A, int U)
{
    int row = blockIdx.x, j = threadIdx.x;
    int b = row / U, u = row - b * U;
    float m   = stats[j] * invN;
    float var = stats[64 + j] * invN - m * m;
    float sc  = rsqrtf(var + 1e-5f) * gamma[j];
    float off = beta[j] - m * sc;
    float s = 0.f;
    for (int a = 0; a < A; ++a)
        s += (float)raw[(((size_t)b * A + a) * U + u) * 64 + j] * sc + off;
    if (o32) o32[row * 64 + j] = s;
    if (o16) o16[row * 64 + j] = (_Float16)s;
}

// eid-masked aggregation with on-the-fly normalization of ap_res/ue_res:
// out = (sumU - apn) + (sumA - uen), f16
__global__ void agg2_k(const _Float16* __restrict__ apr, const float* __restrict__ stA,
                       const float* __restrict__ gA, const float* __restrict__ bA,
                       const _Float16* __restrict__ uer, const float* __restrict__ stB,
                       const float* __restrict__ gB, const float* __restrict__ bB,
                       const float* __restrict__ sap, const float* __restrict__ sue,
                       _Float16* __restrict__ out, float invN, int A, int U, int total)
{
    int idx = blockIdx.x * 256 + threadIdx.x;
    if (idx >= total) return;
    int j = idx & 63, e = idx >> 6;
    float mA = stA[j] * invN, vA = stA[64 + j] * invN - mA * mA;
    float scA = rsqrtf(vA + 1e-5f) * gA[j], oA = bA[j] - mA * scA;
    float mB = stB[j] * invN, vB = stB[64 + j] * invN - mB * mB;
    float scB = rsqrtf(vB + 1e-5f) * gB[j], oB = bB[j] - mB * scB;
    float apn = (float)apr[idx] * scA + oA;
    float uen = (float)uer[idx] * scB + oB;
    int ap = e / U;
    int b  = e / (A * U);
    int u  = e - (e / U) * U;
    float v = sap[ap * 64 + j] - apn + sue[(b * U + u) * 64 + j] - uen;
    out[idx] = (_Float16)v;
}

// post: sigmoid(hE . w + b)
__global__ void post_k(const _Float16* __restrict__ hE, const float* __restrict__ w,
                       const float* __restrict__ b, float* __restrict__ out, int N)
{
    int e = blockIdx.x * 256 + threadIdx.x;
    if (e >= N) return;
    float s = b[0];
    const _Float16* p = hE + (size_t)e * 64;
#pragma unroll
    for (int j = 0; j < 64; ++j) s += (float)p[j] * w[j];
    out[e] = 1.f / (1.f + expf(-s));
}

__global__ void cvt_k(const float* __restrict__ s, _Float16* __restrict__ d, int n)
{
    int i = blockIdx.x * 256 + threadIdx.x;
    if (i < n) d[i] = (_Float16)s[i];
}

// ---------------------------------------------------------------------------
// Host orchestration
// ---------------------------------------------------------------------------
extern "C" void kernel_launch(void* const* d_in, const int* in_sizes, int n_in,
                              void* d_out, int out_size, void* d_ws, size_t ws_size,
                              hipStream_t stream)
{
    (void)in_sizes; (void)out_size;
    const int Ne = B_ * A_ * U_;   // 262144
    const int Na = B_ * A_;        // 4096
    const int Nu = B_ * U_;        // 8192
    if (n_in < 74) return;

    auto F = [&](int i) { return (const float*)d_in[i]; };

    // params layout (tree_leaves, dict keys sorted):
    // 0..3 : ap_feat, ue_feat, e_a2u, e_u2a
    // 4,5  : post.W, post.b
    // 6..9 : pre_ap.W,b,beta,gamma   10..13: pre_edge.*   14..17: pre_ue.*
    // 18+8k: u(k+1).W1,W2,W3,b1,b2,b3,beta,gamma
    struct UL { const float *W1, *W2, *W3, *b1, *b2, *b3, *beta, *gamma; };
    UL ul[7];
    for (int k = 0; k < 7; ++k) {
        int o = 18 + 8 * k;
        ul[k] = { F(o), F(o + 1), F(o + 2), F(o + 3), F(o + 4), F(o + 5), F(o + 6), F(o + 7) };
    }

    // workspace carve
    char* ws = (char*)d_ws;
    size_t off = 0;
    auto alloc = [&](size_t bytes) -> void* {
        void* p = ws + off;
        off = (off + bytes + 255) & ~(size_t)255;
        return p;
    };
    _Float16* hE   = (_Float16*)alloc((size_t)Ne * 64 * 2);
    _Float16* hE2  = (_Float16*)alloc((size_t)Ne * 64 * 2);
    _Float16* hEn  = (_Float16*)alloc((size_t)Ne * 64 * 2);
    _Float16* hAGG = (_Float16*)alloc((size_t)Ne * 64 * 2);
    _Float16* hAP  = (_Float16*)alloc((size_t)Na * 64 * 2);
    _Float16* hAPn = (_Float16*)alloc((size_t)Na * 64 * 2);
    _Float16* hUE  = (_Float16*)alloc((size_t)Nu * 64 * 2);
    _Float16* hUEn = (_Float16*)alloc((size_t)Nu * 64 * 2);
    _Float16* Sa16 = (_Float16*)alloc((size_t)Na * 64 * 2);
    _Float16* Su16 = (_Float16*)alloc((size_t)Nu * 64 * 2);
    _Float16* RAW  = (_Float16*)alloc((size_t)Ne * 64 * 2);   // f16 pre-BN raw
    _Float16* C2   = (_Float16*)alloc((size_t)Ne * 64 * 2);   // f16 pre-BN raw
    float* sAPf = (float*)alloc((size_t)Na * 64 * 4);
    float* sUEf = (float*)alloc((size_t)Nu * 64 * 4);
    float* partialBuf = (float*)alloc((size_t)(Ne / 128) * 128 * 4);  // 1 MB
    float* statsA = (float*)alloc(128 * 4);
    float* statsB = (float*)alloc(128 * 4);
    _Float16 *w1h[7], *w2h[7], *w3h[7];
    for (int k = 0; k < 7; ++k) {
        w1h[k] = (_Float16*)alloc(8192 * 2);
        w2h[k] = (_Float16*)alloc(4096 * 2);
        w3h[k] = (_Float16*)alloc(4096 * 2);
    }
    if (off > ws_size) return;  // insufficient workspace

    // convert weights to f16 once
    for (int k = 0; k < 7; ++k) {
        cvt_k<<<32, 256, 0, stream>>>(ul[k].W1, w1h[k], 8192);
        cvt_k<<<16, 256, 0, stream>>>(ul[k].W2, w2h[k], 4096);
        cvt_k<<<16, 256, 0, stream>>>(ul[k].W3, w3h[k], 4096);
    }

    // fused GEMM (+partial stats) followed by deterministic stats combine
    auto run_gemm = [&](int k, const _Float16* s0, int m0, const _Float16* s1, int m1,
                        int Nrows, _Float16* rawout, float* stats) {
        mlp3_wmma<<<Nrows / 128, 256, 0, stream>>>(s0, m0, s1, m1,
                                                   w1h[k], ul[k].b1, w2h[k], ul[k].b2,
                                                   w3h[k], ul[k].b3, rawout, partialBuf,
                                                   A_, U_);
        stats2_k<<<1, 1024, 0, stream>>>(partialBuf, Nrows / 128, stats);
    };
    auto run_norm = [&](const _Float16* raw, const float* stats,
                        const float* gamma, const float* beta,
                        int Nrows, _Float16* o16) {
        int total = Nrows * 64;
        norm_k<<<(total + 255) / 256, 256, 0, stream>>>(raw, stats, gamma, beta,
                                                        1.0f / (float)Nrows, o16, total);
    };

    const float invNe = 1.0f / (float)Ne;

    // ---- PreLayer ----
    pre1_k<<<Na / 128, 256, 0, stream>>>(F(0), 1, F(6), F(7), RAW, partialBuf);
    stats2_k<<<1, 1024, 0, stream>>>(partialBuf, Na / 128, statsA);
    run_norm(RAW, statsA, F(9), F(8), Na, hAP);                    // pre_ap

    pre1_k<<<Nu / 128, 256, 0, stream>>>(F(1), 1, F(14), F(15), RAW, partialBuf);
    stats2_k<<<1, 1024, 0, stream>>>(partialBuf, Nu / 128, statsA);
    run_norm(RAW, statsA, F(17), F(16), Nu, hUE);                  // pre_ue

    pre1_k<<<Ne / 128, 256, 0, stream>>>(F(2), 2, F(10), F(11), RAW, partialBuf);
    stats2_k<<<1, 1024, 0, stream>>>(partialBuf, Ne / 128, statsA);
    run_norm(RAW, statsA, F(13), F(12), Ne, hE);                   // pre_edge (a2u)

    pre1_k<<<Ne / 128, 256, 0, stream>>>(F(3), 2, F(10), F(11), RAW, partialBuf);
    stats2_k<<<1, 1024, 0, stream>>>(partialBuf, Ne / 128, statsA);
    run_norm(RAW, statsA, F(13), F(12), Ne, hE2);                  // pre_edge (u2a)

    // ---- 2 iterations of shared UpdateLayer ----
    for (int it = 0; it < 2; ++it) {
        const _Float16 *E, *E2, *AP, *UE;
        _Float16 *Eout, *APout, *UEout;
        if (it == 0) { E = hE;  E2 = hE2; AP = hAP;  UE = hUE;
                       Eout = hEn; APout = hAPn; UEout = hUEn; }
        else         { E = hEn; E2 = hEn; AP = hAPn; UE = hUEn;   // hid_e2 == hid_e after iter 0
                       Eout = hE;  APout = hAP;  UEout = hUE; }

        // APConv: m_ap = BN(MLP(cat[e2, ue_b], u1)); fused normalize+sum over U
        run_gemm(0, E2, 0, UE, 2, Ne, RAW, statsA);
        normsumU_k<<<Na, 64, 0, stream>>>(RAW, statsA, ul[0].gamma, ul[0].beta,
                                          invNe, nullptr, Sa16, U_);
        // new_ap = BN(MLP(cat[ap, m_ap.sum], u2))
        run_gemm(1, AP, 0, Sa16, 0, Na, RAW, statsA);
        run_norm(RAW, statsA, ul[1].gamma, ul[1].beta, Na, APout);

        // UEConv: m_ue = BN(MLP(cat[e, ap_b], u3)); fused normalize+sum over A
        run_gemm(2, E, 0, AP, 1, Ne, RAW, statsA);
        normsumA_k<<<Nu, 64, 0, stream>>>(RAW, statsA, ul[2].gamma, ul[2].beta,
                                          invNe, nullptr, Su16, A_, U_);
        // new_ue = BN(MLP(cat[ue, m_ue.sum], u4))
        run_gemm(3, UE, 0, Su16, 0, Nu, RAW, statsA);
        run_norm(RAW, statsA, ul[3].gamma, ul[3].beta, Nu, UEout);

        // EdgeConv: ap_res raw -> RAW (+statsA), ue_res raw -> C2 (+statsB)
        run_gemm(4, AP, 1, E2, 0, Ne, RAW, statsA);
        run_gemm(5, UE, 2, E, 0, Ne, C2, statsB);
        normsumU_k<<<Na, 64, 0, stream>>>(RAW, statsA, ul[4].gamma, ul[4].beta,
                                          invNe, sAPf, nullptr, U_);
        normsumA_k<<<Nu, 64, 0, stream>>>(C2, statsB, ul[5].gamma, ul[5].beta,
                                          invNe, sUEf, nullptr, A_, U_);
        agg2_k<<<(Ne * 64 + 255) / 256, 256, 0, stream>>>(
            RAW, statsA, ul[4].gamma, ul[4].beta,
            C2, statsB, ul[5].gamma, ul[5].beta,
            sAPf, sUEf, hAGG, invNe, A_, U_, Ne * 64);

        // new_e = BN(MLP(cat[e, agg], u7))
        run_gemm(6, E, 0, hAGG, 0, Ne, RAW, statsA);
        run_norm(RAW, statsA, ul[6].gamma, ul[6].beta, Ne, Eout);
    }

    // ---- post: sigmoid(hid_e @ W.T + b) ----
    post_k<<<(Ne + 255) / 256, 256, 0, stream>>>(hE, F(4), F(5), (float*)d_out, Ne);
}
